// GraphAttentionLayer_83975200571944
// MI455X (gfx1250) — matrix operand order repaired and checked by reading, not compiled, not run on hardware
//
#include <hip/hip_runtime.h>
#include <hip/hip_bf16.h>

typedef __attribute__((ext_vector_type(16))) __bf16 v16bf;
typedef __attribute__((ext_vector_type(8)))  __bf16 v8bf;
typedef __attribute__((ext_vector_type(8)))  float  v8f;

#define BB 8
#define NN 2048
#define FF 128

// -------- CDNA5 async global->LDS copy (ASYNCcnt) with sync fallback --------
#if defined(__AMDGCN__) && \
    __has_builtin(__builtin_amdgcn_global_load_async_to_lds_b128) && \
    __has_builtin(__builtin_amdgcn_s_wait_asynccnt)
#define GAT_ASYNC 1
#else
#define GAT_ASYNC 0
#endif

typedef int v4i __attribute__((ext_vector_type(4)));
typedef __attribute__((address_space(1))) v4i as1_v4i;  // global
typedef __attribute__((address_space(3))) v4i as3_v4i;  // LDS

__device__ __forceinline__ void gat_async_b128(const void* gsrc, void* ldst) {
#if GAT_ASYNC
    // generic shared pointers carry the LDS byte offset in addr[31:0]
    __builtin_amdgcn_global_load_async_to_lds_b128(
        (as1_v4i*)(uintptr_t)gsrc,
        (as3_v4i*)(unsigned int)(uintptr_t)ldst, 0, 0);
#else
    (void)gsrc; (void)ldst;
#endif
}

// ---------------------------------------------------------------------------
// Kernel 0: pre-swizzle W (f32 [FIN][FOUT]) into WMMA B-fragment layout, bf16.
// B-matrix 32x16 (16-bit): lane holds N = lane&15; element e holds
// K = (e<8 ? e : e+8) + (lane<16 ? 0 : 8).  Layout: wfrag[(kt*8+nt)*32 + lane][16].
// ---------------------------------------------------------------------------
__global__ void gat_prep_wfrag(const float* __restrict__ W,
                               __bf16* __restrict__ wfrag) {
    const int t = blockIdx.x * blockDim.x + threadIdx.x;
    if (t >= 4 * 8 * 32) return;
    const int lane = t & 31;
    const int nt   = (t >> 5) & 7;
    const int kt   = t >> 8;
    const int col  = nt * 16 + (lane & 15);
    const int kb   = kt * 32 + ((lane & 16) ? 8 : 0);
    __bf16* dst = wfrag + t * 16;
#pragma unroll
    for (int e = 0; e < 16; ++e)
        dst[e] = (__bf16)W[(kb + (e < 8 ? e : e + 8)) * FF + col];
}

// ---------------------------------------------------------------------------
// Kernel 1: h = x @ W via v_wmma_f32_16x16x32_bf16.
// One wave = 16 rows x 128 cols.  Writes h transposed: h_t[b][fout][n] (bf16)
// so that the attention GEMM's B-fragments are contiguous in j.
// A-matrix 16x32 (16-bit): lane holds M = lane&15; element e holds
// K = (e<8 ? e : e+8) + (lane<16 ? 0 : 8).
// ---------------------------------------------------------------------------
__global__ void __launch_bounds__(256)
gat_h_gemm(const float* __restrict__ x,
           const __bf16* __restrict__ wfrag,
           __bf16* __restrict__ h_t) {
    const int lane = threadIdx.x & 31;
    const int wv   = threadIdx.x >> 5;
    const int tile = blockIdx.x * 8 + wv;     // [0, 1024) row tiles of 16
    const int g0   = tile << 4;               // global row base (B*N rows)
    const int bIdx = g0 >> 11;
    const int n0   = g0 & (NN - 1);
    const int row  = lane & 15;
    const int hf   = lane >> 4;

    v8f c[8] = {};
    const float* xrow = x + (size_t)(g0 + row) * FF;
#pragma unroll
    for (int kt = 0; kt < 4; ++kt) {
        const int kb = kt * 32 + hf * 8;
        float4 r0 = *(const float4*)(xrow + kb);
        float4 r1 = *(const float4*)(xrow + kb + 4);
        float4 r2 = *(const float4*)(xrow + kb + 16);
        float4 r3 = *(const float4*)(xrow + kb + 20);
        v16bf a;
        a[0]=(__bf16)r0.x;  a[1]=(__bf16)r0.y;  a[2]=(__bf16)r0.z;  a[3]=(__bf16)r0.w;
        a[4]=(__bf16)r1.x;  a[5]=(__bf16)r1.y;  a[6]=(__bf16)r1.z;  a[7]=(__bf16)r1.w;
        a[8]=(__bf16)r2.x;  a[9]=(__bf16)r2.y;  a[10]=(__bf16)r2.z; a[11]=(__bf16)r2.w;
        a[12]=(__bf16)r3.x; a[13]=(__bf16)r3.y; a[14]=(__bf16)r3.z; a[15]=(__bf16)r3.w;
        const v16bf* wp = (const v16bf*)(wfrag + (size_t)kt * 8 * 32 * 16);
#pragma unroll
        for (int nt = 0; nt < 8; ++nt) {
            v16bf bm = wp[nt * 32 + lane];
            c[nt] = __builtin_amdgcn_wmma_f32_16x16x32_bf16(
                false, a, false, bm, (short)0, c[nt], false, false);
        }
    }
    // D layout: VGPR r -> M = r + (lane<16?0:8), N(col) = lane&15.
    // Transposed store: fixed fout per lane, 8 consecutive n -> one 16B store.
#pragma unroll
    for (int nt = 0; nt < 8; ++nt) {
        const int fout = nt * 16 + row;
        v8bf pk;
#pragma unroll
        for (int r = 0; r < 8; ++r) pk[r] = (__bf16)c[nt][r];
        *(v8bf*)(h_t + (size_t)(bIdx * FF + fout) * NN + n0 + hf * 8) = pk;
    }
}

// ---------------------------------------------------------------------------
// Kernel 2: s_i = h . a_src, s_j = h . a_dst  (from transposed bf16 h).
// ---------------------------------------------------------------------------
__global__ void __launch_bounds__(256)
gat_scores(const __bf16* __restrict__ h_t,
           const float* __restrict__ a_src,
           const float* __restrict__ a_dst,
           float* __restrict__ s_src,
           float* __restrict__ s_dst) {
    const int t = blockIdx.x * blockDim.x + threadIdx.x;   // [0, B*N)
    const int b = t >> 11, n = t & (NN - 1);
    const __bf16* hp = h_t + (size_t)b * FF * NN + n;
    float ss = 0.f, sd = 0.f;
#pragma unroll 4
    for (int f = 0; f < FF; ++f) {
        const float hv = (float)hp[(size_t)f * NN];
        ss += hv * a_src[f];
        sd += hv * a_dst[f];
    }
    s_src[t] = ss;
    s_dst[t] = sd;
}

// ---------------------------------------------------------------------------
// Kernel 3: fused masked-softmax attention + out = attn @ h (flash-style).
// Block = 8 waves, each wave owns 16 rows (i) and the full FOUT=128 (8 C-frags).
// The per-j-tile h slab is double-buffered in LDS and streamed with
// GLOBAL_LOAD_ASYNC_TO_LDS_B128 (ASYNCcnt) so the copy of tile jt+1 overlaps
// the softmax+WMMA work on tile jt; one barrier per tile.
// ---------------------------------------------------------------------------
__global__ void __launch_bounds__(256)
gat_attn(const __bf16* __restrict__ h_t,
         const float* __restrict__ s_src,
         const float* __restrict__ s_dst,
         const int* __restrict__ adj,
         float* __restrict__ out) {
    __shared__ __align__(32) __bf16 lds_h[2][FF * 32];  // 2 x 8KB h slab
    __shared__ __align__(16) float  lds_sj[2][32];
    __shared__ __align__(16) float  lds_bc[8 * 16];     // per-wave row bcast

    const int tid   = threadIdx.x;
    const int lane  = tid & 31;
    const int wv    = tid >> 5;
    const int b     = blockIdx.x >> 4;
    const int ibase = (blockIdx.x & 15) * 128 + wv * 16;
    const int row   = lane & 15;
    const int hf    = lane >> 4;
    const int i     = ibase + row;

    const float sI = s_src[b * NN + i];
    const int* adjrow = adj + (size_t)i * NN;

    // staging lambda: copy h_t[b][0:128][jbase:jbase+32] + s_dst tile -> buf
    const int sf = tid >> 1;               // feature row this thread stages
    const int sq = (tid & 1) * 16;         // which 16-element half of the row
    auto stage = [&](int jbase, int buf) {
#if GAT_ASYNC
        const char* gsrc = (const char*)(h_t + (size_t)(b * FF + sf) * NN + jbase + sq);
        char*       ldst = (char*)(&lds_h[buf][sf * 32 + sq]);
        gat_async_b128(gsrc,      ldst);
        gat_async_b128(gsrc + 16, ldst + 16);
        if (tid < 8)
            gat_async_b128(s_dst + b * NN + jbase + tid * 4, &lds_sj[buf][tid * 4]);
#else
        v16bf hv = *(const v16bf*)(h_t + (size_t)(b * FF + sf) * NN + jbase + sq);
        *(v16bf*)(&lds_h[buf][sf * 32 + sq]) = hv;
        if (tid < 32) lds_sj[buf][tid] = s_dst[b * NN + jbase + tid];
#endif
    };

    v8f c[8] = {};
    float m = -1e30f, lsum = 0.f;
    int cnt = 0;

    stage(0, 0);
    for (int jt = 0; jt < NN / 32; ++jt) {
        const int buf   = jt & 1;
        const int jbase = jt * 32;
#if GAT_ASYNC
        __builtin_amdgcn_s_wait_asynccnt(0);   // own async writes done
#endif
        __syncthreads();                       // everyone's writes visible
        if (jt + 1 < NN / 32) stage(jbase + 32, buf ^ 1);

        if (jt + 1 < NN / 32) __builtin_prefetch(adjrow + jbase + 32, 0, 1);

        // Build e-block with A-fragment K mapping: two contiguous runs of 8.
        const int k0 = hf * 8;
        int4 a0 = *(const int4*)(adjrow + jbase + k0);
        int4 a1 = *(const int4*)(adjrow + jbase + k0 + 4);
        int4 a2 = *(const int4*)(adjrow + jbase + k0 + 16);
        int4 a3 = *(const int4*)(adjrow + jbase + k0 + 20);
        float4 j0 = *(const float4*)(&lds_sj[buf][k0]);
        float4 j1 = *(const float4*)(&lds_sj[buf][k0 + 4]);
        float4 j2 = *(const float4*)(&lds_sj[buf][k0 + 16]);
        float4 j3 = *(const float4*)(&lds_sj[buf][k0 + 20]);

        const int   ms[16] = {a0.x,a0.y,a0.z,a0.w, a1.x,a1.y,a1.z,a1.w,
                              a2.x,a2.y,a2.z,a2.w, a3.x,a3.y,a3.z,a3.w};
        const float sj[16] = {j0.x,j0.y,j0.z,j0.w, j1.x,j1.y,j1.z,j1.w,
                              j2.x,j2.y,j2.z,j2.w, j3.x,j3.y,j3.z,j3.w};

        float ev[16];
        float mb = -1e30f;
        int cb = 0;
#pragma unroll
        for (int e = 0; e < 16; ++e) {
            float v = sI + sj[e];
            v = (v > 0.f) ? v : 0.2f * v;          // LeakyReLU(0.2)
            const bool on = (ms[e] != 0);
            ev[e] = on ? v : -1e30f;               // masked -> -inf-ish
            cb += on ? 1 : 0;
            mb = fmaxf(mb, ev[e]);
        }
        // row stats live on the lane pair (lane, lane^16) sharing row M
        mb = fmaxf(mb, __shfl_xor(mb, 16, 32));
        const float mnew  = fmaxf(m, mb);
        const float alpha = __expf(m - mnew);
        m = mnew;

        v16bf pa;
        float sb = 0.f;
#pragma unroll
        for (int e = 0; e < 16; ++e) {
            const float p = __expf(ev[e] - mnew);  // exp(-huge) == 0, no NaN
            sb += p;
            pa[e] = (__bf16)p;
        }
        sb += __shfl_xor(sb, 16, 32);
        cb += __shfl_xor(cb, 16, 32);
        lsum = lsum * alpha + sb;
        cnt += cb;

        // broadcast alpha per row to the C-fragment row layout (same-wave DS
        // ops are in-order; no barrier needed)
        if (lane < 16) lds_bc[wv * 16 + lane] = alpha;
        float4 al0 = *(const float4*)(lds_bc + wv * 16 + hf * 8);
        float4 al1 = *(const float4*)(lds_bc + wv * 16 + hf * 8 + 4);
        const float ar[8] = {al0.x,al0.y,al0.z,al0.w, al1.x,al1.y,al1.z,al1.w};
#pragma unroll
        for (int ct = 0; ct < 8; ++ct)
#pragma unroll
            for (int r = 0; r < 8; ++r) c[ct][r] *= ar[r];

        // accumulate: out_tile += P(16x32) @ h(32x128)
#pragma unroll
        for (int ct = 0; ct < 8; ++ct) {
            v8bf blo = *(const v8bf*)(&lds_h[buf][(ct * 16 + row) * 32 + hf * 8]);
            v8bf bhi = *(const v8bf*)(&lds_h[buf][(ct * 16 + row) * 32 + 16 + hf * 8]);
            v16bf bm = __builtin_shufflevector(blo, bhi, 0,1,2,3,4,5,6,7,8,9,10,11,12,13,14,15);
            c[ct] = __builtin_amdgcn_wmma_f32_16x16x32_bf16(
                false, pa, false, bm, (short)0, c[ct], false, false);
        }
    }

    // finalize: rows with no neighbors -> 0 (matches nan_to_num of softmax)
    const float inv = (cnt > 0) ? (1.f / lsum) : 0.f;
    if (lane < 16) lds_bc[wv * 16 + lane] = inv;
    float4 iv0 = *(const float4*)(lds_bc + wv * 16 + hf * 8);
    float4 iv1 = *(const float4*)(lds_bc + wv * 16 + hf * 8 + 4);
    const float ir[8] = {iv0.x,iv0.y,iv0.z,iv0.w, iv1.x,iv1.y,iv1.z,iv1.w};
#pragma unroll
    for (int ct = 0; ct < 8; ++ct)
#pragma unroll
        for (int r = 0; r < 8; ++r)
            out[(size_t)(b * NN + ibase + hf * 8 + r) * FF + ct * 16 + row] =
                c[ct][r] * ir[r];
}

// ---------------------------------------------------------------------------
extern "C" void kernel_launch(void* const* d_in, const int* in_sizes, int n_in,
                              void* d_out, int out_size, void* d_ws, size_t ws_size,
                              hipStream_t stream) {
    (void)in_sizes; (void)n_in; (void)out_size; (void)ws_size;
    const float* x     = (const float*)d_in[0];
    const int*   adj   = (const int*)d_in[1];
    const float* W     = (const float*)d_in[2];
    const float* a_src = (const float*)d_in[3];
    const float* a_dst = (const float*)d_in[4];
    float* out = (float*)d_out;

    char* ws = (char*)d_ws;
    __bf16* h_t   = (__bf16*)(ws);                              // 4 MiB
    __bf16* wfrag = (__bf16*)(ws + (4u << 20));                 // 32 KiB
    float*  s_src = (float*) (ws + (4u << 20) + (32u << 10));   // 64 KiB
    float*  s_dst = (float*) (ws + (4u << 20) + (96u << 10));   // 64 KiB

    gat_prep_wfrag<<<4,   256, 0, stream>>>(W, wfrag);
    gat_h_gemm    <<<128, 256, 0, stream>>>(x, wfrag, h_t);
    gat_scores    <<<64,  256, 0, stream>>>(h_t, a_src, a_dst, s_src, s_dst);
    gat_attn      <<<128, 256, 0, stream>>>(h_t, s_src, s_dst, adj, out);
}
